// GNNModule_E1_19138374271366
// MI455X (gfx1250) — matrix operand order repaired and checked by reading, not compiled, not run on hardware
//
#include <hip/hip_runtime.h>
#include <hip/hip_bf16.h>
#include <math.h>

// ---------------- problem constants ----------------
#define H     128
#define HEADS 4
#define NN    25000          // nodes
#define NE    400000         // edges
#define NPAD  25008          // nodes padded to multiple of 16
#define EPS   1e-5f

typedef __attribute__((ext_vector_type(16))) __bf16 v16bf;
typedef __attribute__((ext_vector_type(8)))  float  v8f;

__device__ __forceinline__ unsigned short f2bf(float f) {
  union { float f; unsigned u; } v; v.f = f;
  unsigned r = v.u + 0x7FFFu + ((v.u >> 16) & 1u);   // round-to-nearest-even
  return (unsigned short)(r >> 16);
}

union FragU { v16bf v; uint4 u[2]; };

// A fragment, 16x32 bf16, row-major source. ISA 7.12.2:
// lane<16: M=lane, K = k0+0..7 (v0-3) and k0+16..23 (v4-7); lane>=16: +8 on K base.
__device__ __forceinline__ v16bf load_frag_a(const unsigned short* A, int lda,
                                             int m0, int k0, int lane) {
  int r  = m0 + (lane & 15);
  int kb = k0 + ((lane >> 4) << 3);
  const unsigned short* p = A + (size_t)r * lda + kb;
  FragU f;
  f.u[0] = *(const uint4*)(p);
  f.u[1] = *(const uint4*)(p + 16);
  return f.v;
}

// B fragment, 32x16 bf16, from Wt[N,K] row-major (i.e. B transposed).
// lane&15 = N, K = k0+0..7 and k0+16..23; lane>=16: +8 on K base (matches A).
__device__ __forceinline__ v16bf load_frag_b(const unsigned short* Bt, int ldb,
                                             int n0, int k0, int lane) {
  int r  = n0 + (lane & 15);
  int kb = k0 + ((lane >> 4) << 3);
  const unsigned short* p = Bt + (size_t)r * ldb + kb;
  FragU f;
  f.u[0] = *(const uint4*)(p);
  f.u[1] = *(const uint4*)(p + 16);
  return f.v;
}

// ---------------- register-resident-B WMMA GEMM ----------------
// C[M, N] = epi( (A[M, K=KSTEPS*32]bf16 @ Bt[N,K]^T + bias) * scale )
// One wave owns NT 16-wide column tiles; B fragments are loaded ONCE into
// registers and reused across a strided loop over 16-row M tiles, so the
// steady state is 2 b128 A-loads feeding NT WMMAs per K-step.
// __launch_bounds__(256, 1): let the allocator use the full wave32 VGPR
// budget so the B fragments do NOT spill to scratch (round-2 failure mode).
// epi: 0=id, 1=relu, 2=gelu(exact). Writes f32 (Cf) and/or bf16 (Cb).
template<int KSTEPS, int NT>
__global__ __launch_bounds__(256, 1)
void gemm_bf16_rb_k(const unsigned short* __restrict__ A, int lda,
                    const unsigned short* __restrict__ Bt, int ldb,
                    const float* __restrict__ bias,
                    float* __restrict__ Cf,
                    unsigned short* __restrict__ Cb, int ldc,
                    int M, int N, int Mtiles, int ngroups,
                    int epi, float scale)
{
  const int lane = threadIdx.x & 31;
  const int gw   = blockIdx.x * (blockDim.x >> 5) + (threadIdx.x >> 5);
  const int totw = gridDim.x * (blockDim.x >> 5);
  const int ngrp = gw % ngroups;                  // column group (wave-uniform)
  const int mt0  = gw / ngroups;                  // first m-tile
  const int mstride = totw / ngroups;             // exact (host guarantees)
  const int n0 = ngrp * (NT * 16);

  // persistent B fragments (weights, L2-hot): loaded once per wave
  v16bf Bf[NT][KSTEPS];
#pragma unroll
  for (int t = 0; t < NT; ++t)
#pragma unroll
    for (int k = 0; k < KSTEPS; ++k)
      Bf[t][k] = load_frag_b(Bt, ldb, n0 + t * 16, k * 32, lane);

  const v8f vzero = {};
  for (int mt = mt0; mt < Mtiles; mt += mstride) {
    const int m0 = mt * 16;
    v8f acc[NT];
#pragma unroll
    for (int t = 0; t < NT; ++t) acc[t] = vzero;

#pragma unroll
    for (int k = 0; k < KSTEPS; ++k) {
      v16bf a = load_frag_a(A, lda, m0, k * 32, lane);
#pragma unroll
      for (int t = 0; t < NT; ++t)
        acc[t] = __builtin_amdgcn_wmma_f32_16x16x32_bf16(false, a, false, Bf[t][k],
                                                         (short)0, acc[t], false, false);
    }

    const int rbase = m0 + ((lane >> 4) << 3);
#pragma unroll
    for (int t = 0; t < NT; ++t) {
      const int col = n0 + t * 16 + (lane & 15);
      const float bv = (bias && col < N) ? bias[col] : 0.f;
#pragma unroll
      for (int r = 0; r < 8; ++r) {
        int row = rbase + r;
        float v = (acc[t][r] + bv) * scale;
        if (epi == 1)      v = fmaxf(v, 0.f);
        else if (epi == 2) v = 0.5f * v * (1.f + erff(v * 0.70710678118f));
        if (row < M && col < N) {
          if (Cf) Cf[(size_t)row * ldc + col] = v;
          if (Cb) Cb[(size_t)row * ldc + col] = f2bf(v);
        }
      }
    }
  }
}

// ---------------- weight prep: Wt[Npad,K] bf16 from W[K,N] f32 ----------------
__global__ void prep_wt_k(const float* __restrict__ W, unsigned short* __restrict__ Wt,
                          int K, int N, int Npad)
{
  int i = blockIdx.x * 256 + threadIdx.x;
  if (i >= Npad * K) return;
  int n = i / K, k = i - n * K;
  float v = (n < N) ? W[(size_t)k * N + n] : 0.f;
  Wt[i] = f2bf(v);
}

// ---------------- gathered edge matrix [E,384] = [nodes[src], hE, nodes[dst]] ----------------
__global__ void build_edge3_k(const float* __restrict__ nodes, const float* __restrict__ hE,
                              const int* __restrict__ src, const int* __restrict__ dst,
                              unsigned short* __restrict__ out, long total)
{
  long i = (long)blockIdx.x * 256 + threadIdx.x;
  if (i >= total) return;
  int  c = (int)(i % 384);
  long e = i / 384;
  float v;
  if (c < 128)        v = nodes[(size_t)src[e] * H + c];
  else if (c < 256)   v = hE[(size_t)e * H + (c - 128)];
  else                v = nodes[(size_t)dst[e] * H + (c - 256)];
  out[i] = f2bf(v);
}

// ---------------- misc elementwise ----------------
__global__ void fill_f32_k(float* p, float v, long n) {
  long i = (long)blockIdx.x * 256 + threadIdx.x;
  if (i < n) p[i] = v;
}
__global__ void cvt_bf_k(const float* __restrict__ in, unsigned short* __restrict__ out, long n) {
  long i = (long)blockIdx.x * 256 + threadIdx.x;
  if (i < n) out[i] = f2bf(in[i]);
}

// ---------------- segment softmax pieces ----------------
__global__ void seg_max_k(const float* __restrict__ lg, const int* __restrict__ src,
                          float* __restrict__ mx, int e4)
{
  int i = blockIdx.x * 256 + threadIdx.x;
  if (i >= e4) return;
  int e = i >> 2, h = i & 3;
  __hip_atomic_fetch_max(&mx[(size_t)src[e] * HEADS + h], lg[i],
                         __ATOMIC_RELAXED, __HIP_MEMORY_SCOPE_AGENT);
}
__global__ void seg_exp_k(float* __restrict__ lg, const int* __restrict__ src,
                          const float* __restrict__ mx, float* __restrict__ den, int e4)
{
  int i = blockIdx.x * 256 + threadIdx.x;
  if (i >= e4) return;
  int e = i >> 2, h = i & 3;
  float v = expf(lg[i] - mx[(size_t)src[e] * HEADS + h]);
  lg[i] = v;
  atomicAdd(&den[(size_t)src[e] * HEADS + h], v);
}
__global__ void seg_agg_k(const float* __restrict__ ex, const float* __restrict__ den,
                          const float* __restrict__ Vm, const int* __restrict__ src,
                          float* __restrict__ agg, long total)
{
  long i = (long)blockIdx.x * 256 + threadIdx.x;
  if (i >= total) return;
  int  c = (int)(i & 127);
  long e = i >> 7;
  int  h = c >> 5;                       // d = H/HEADS = 32
  int  s = src[e];
  float at = ex[e * HEADS + h] / den[(size_t)s * HEADS + h];
  atomicAdd(&agg[(size_t)s * H + c], at * Vm[i]);
}

// ---------------- BatchNorm (training-mode batch stats), C = 128 ----------------
__global__ void bn_reduce_k(const float* __restrict__ P, const float* __restrict__ Q,
                            int M, float* __restrict__ mv)
{
  int c = blockIdx.x;                    // one block per column
  float s = 0.f, s2 = 0.f;
  for (int i = threadIdx.x; i < M; i += 256) {
    float x = P[(size_t)i * H + c];
    if (Q) x += Q[(size_t)i * H + c];
    s += x; s2 += x * x;
  }
  __shared__ float sh0[256], sh1[256];
  sh0[threadIdx.x] = s; sh1[threadIdx.x] = s2;
  __syncthreads();
  for (int st = 128; st > 0; st >>= 1) {
    if (threadIdx.x < st) {
      sh0[threadIdx.x] += sh0[threadIdx.x + st];
      sh1[threadIdx.x] += sh1[threadIdx.x + st];
    }
    __syncthreads();
  }
  if (threadIdx.x == 0) {
    float m = sh0[0] / (float)M;
    mv[c]     = m;
    mv[H + c] = sh1[0] / (float)M - m * m;   // biased variance
  }
}
__global__ void bn_norm_k(const float* __restrict__ P, const float* __restrict__ Q,
                          const float* __restrict__ mv,
                          const float* __restrict__ g, const float* __restrict__ b,
                          float* __restrict__ outF, unsigned short* __restrict__ outB,
                          long total)
{
  long i = (long)blockIdx.x * 256 + threadIdx.x;
  if (i >= total) return;
  int c = (int)(i & 127);
  float x = P[i];
  if (Q) x += Q[i];
  float y = (x - mv[c]) * rsqrtf(mv[H + c] + EPS) * g[c] + b[c];
  if (outF) outF[i] = y;
  if (outB) outB[i] = f2bf(y);
}

// ---------------- host launch ----------------
extern "C" void kernel_launch(void* const* d_in, const int* in_sizes, int n_in,
                              void* d_out, int out_size, void* d_ws, size_t ws_size,
                              hipStream_t stream) {
  const float* h_V = (const float*)d_in[0];
  const float* h_E = (const float*)d_in[1];
  const int*   ei  = (const int*)d_in[2];
  const int* src = ei;          // edge_idx[0]
  const int* dst = ei + NE;     // edge_idx[1]
  // params in setup_inputs dict order (W,b pairs flattened)
  int p = 4;
  const float *V1W=(const float*)d_in[p++], *V1b=(const float*)d_in[p++];
  const float *V2W=(const float*)d_in[p++], *V2b=(const float*)d_in[p++];
  const float *V3W=(const float*)d_in[p++], *V3b=(const float*)d_in[p++];
  const float *B1W=(const float*)d_in[p++], *B1b=(const float*)d_in[p++];
  const float *B2W=(const float*)d_in[p++], *B2b=(const float*)d_in[p++];
  const float *B3W=(const float*)d_in[p++], *B3b=(const float*)d_in[p++];
  const float *WOW=(const float*)d_in[p++];
  const float *D1W=(const float*)d_in[p++], *D1b=(const float*)d_in[p++];
  const float *D2W=(const float*)d_in[p++], *D2b=(const float*)d_in[p++];
  const float *W11W=(const float*)d_in[p++], *W11b=(const float*)d_in[p++];
  const float *W12W=(const float*)d_in[p++], *W12b=(const float*)d_in[p++];
  const float *W13W=(const float*)d_in[p++], *W13b=(const float*)d_in[p++];
  const float *N0g=(const float*)d_in[p++], *N0b=(const float*)d_in[p++];
  const float *N1g=(const float*)d_in[p++], *N1b=(const float*)d_in[p++];
  const float *N2g=(const float*)d_in[p++], *N2b=(const float*)d_in[p++];

  float* outV = (float*)d_out;                  // [NN,128]
  float* outE = outV + (size_t)NN * H;          // [NE,128]

  // ---- workspace carve ----
  size_t off = 0;
  auto ALLOC = [&](size_t bytes) -> void* {
    void* q = (char*)d_ws + off;
    off = (off + bytes + 255) & ~(size_t)255;
    return q;
  };
  typedef unsigned short u16;
  u16* V1t  = (u16*)ALLOC((size_t)128*256*2);
  u16* V2t  = (u16*)ALLOC((size_t)128*128*2);
  u16* V3t  = (u16*)ALLOC((size_t)128*128*2);
  u16* B1t  = (u16*)ALLOC((size_t)128*384*2);
  u16* B2t  = (u16*)ALLOC((size_t)128*128*2);
  u16* B3t  = (u16*)ALLOC((size_t)16*128*2);
  u16* WOt  = (u16*)ALLOC((size_t)128*128*2);
  u16* D1t  = (u16*)ALLOC((size_t)512*128*2);
  u16* D2t  = (u16*)ALLOC((size_t)128*512*2);
  u16* W11t = (u16*)ALLOC((size_t)128*384*2);
  u16* W12t = (u16*)ALLOC((size_t)128*128*2);
  u16* W13t = (u16*)ALLOC((size_t)128*128*2);

  u16*   M1   = (u16*)ALLOC((size_t)NE*384*2);    // gathered edge activations (bf16)
  u16*   eb1  = (u16*)ALLOC((size_t)NE*128*2);    // edge intermediate A
  u16*   eb2  = (u16*)ALLOC((size_t)NE*128*2);    // edge intermediate B
  float* Vm   = (float*)ALLOC((size_t)NE*128*4);  // V matrix / msg (f32)
  float* lg   = (float*)ALLOC((size_t)NE*HEADS*4);// logits -> exp
  float* mx   = (float*)ALLOC((size_t)NN*HEADS*4);
  float* den  = (float*)ALLOC((size_t)NN*HEADS*4);
  float* agg  = (float*)ALLOC((size_t)NPAD*H*4);
  u16*   aggb = (u16*)ALLOC((size_t)NPAD*H*2);
  float* dh   = (float*)ALLOC((size_t)NPAD*H*4);  // dh then dh2
  float* hVa  = (float*)ALLOC((size_t)NPAD*H*4);
  u16*   hVab = (u16*)ALLOC((size_t)NPAD*H*2);
  u16*   d1b_ = (u16*)ALLOC((size_t)NPAD*512*2);
  float* mv0  = (float*)ALLOC(2*H*4);
  float* mv1  = (float*)ALLOC(2*H*4);
  float* mv2  = (float*)ALLOC(2*H*4);
  (void)ws_size; (void)in_sizes; (void)n_in; (void)out_size;

  auto blks = [](long n) { return (unsigned)((n + 255) / 256); };

  // ---- 0: weight transposes ----
  auto prep = [&](const float* W, u16* Wt, int K, int N, int Np) {
    prep_wt_k<<<blks((long)Np*K), 256, 0, stream>>>(W, Wt, K, N, Np);
  };
  prep(V1W, V1t, 256, 128, 128);  prep(V2W, V2t, 128, 128, 128);  prep(V3W, V3t, 128, 128, 128);
  prep(B1W, B1t, 384, 128, 128);  prep(B2W, B2t, 128, 128, 128);  prep(B3W, B3t, 128, 4, 16);
  prep(WOW, WOt, 128, 128, 128);
  prep(D1W, D1t, 128, 512, 512);  prep(D2W, D2t, 512, 128, 128);
  prep(W11W, W11t, 384, 128, 128); prep(W12W, W12t, 128, 128, 128); prep(W13W, W13t, 128, 128, 128);

  // dispatch: choose (KSTEPS, NT) so persistent-B + acc stays within the
  // directly-addressable 256 VGPRs (no spills).
  // ngroups = Npad / (16*NT); ngroups in {1,2,4,8} always divides waves/block (8).
  auto gemm = [&](const u16* A, int lda, const u16* Bt, int ldb, const float* bias,
                  float* Cf, u16* Cb, int ldc, int M, int Mtiles,
                  int N, int Npadv, int K, int epi, float scale, int blocks) {
    dim3 g((unsigned)blocks); dim3 b(256);
    int ks = K / 32;
    if (ks == 4 && Npadv >= 64) {
      int ng = Npadv / 64;
      gemm_bf16_rb_k<4,4><<<g,b,0,stream>>>(A,lda,Bt,ldb,bias,Cf,Cb,ldc,M,N,Mtiles,ng,epi,scale);
    } else if (ks == 4) {
      int ng = Npadv / 16;
      gemm_bf16_rb_k<4,1><<<g,b,0,stream>>>(A,lda,Bt,ldb,bias,Cf,Cb,ldc,M,N,Mtiles,ng,epi,scale);
    } else if (ks == 8) {
      int ng = Npadv / 32;
      gemm_bf16_rb_k<8,2><<<g,b,0,stream>>>(A,lda,Bt,ldb,bias,Cf,Cb,ldc,M,N,Mtiles,ng,epi,scale);
    } else if (ks == 12) {
      int ng = Npadv / 32;
      gemm_bf16_rb_k<12,2><<<g,b,0,stream>>>(A,lda,Bt,ldb,bias,Cf,Cb,ldc,M,N,Mtiles,ng,epi,scale);
    } else { // ks == 16
      int ng = Npadv / 16;
      gemm_bf16_rb_k<16,1><<<g,b,0,stream>>>(A,lda,Bt,ldb,bias,Cf,Cb,ldc,M,N,Mtiles,ng,epi,scale);
    }
  };
  const int ET = NE / 16;        // 25000 edge row tiles
  const int NT_ = NPAD / 16;     // 1563 node row tiles
  const int EB = 1024;           // blocks for edge GEMMs (8192 waves)
  const int NB = 192;            // blocks for node GEMMs (1536 waves)
  const float isd = 0.17677669529663689f;   // 1/sqrt(d), d=32

  // ---- 1: gather [h_V[src], h_E, h_V[dst]] -> M1 bf16 ----
  build_edge3_k<<<blks((long)NE*384), 256, 0, stream>>>(h_V, h_E, src, dst, M1, (long)NE*384);

  // ---- 2: bias path: relu(relu(M1@B1)@B2)@B3 / sqrt(d) -> logits ----
  gemm(M1, 384, B1t, 384, B1b, nullptr, eb1, 128, NE, ET, 128, 128, 384, 1, 1.f, EB);
  gemm(eb1, 128, B2t, 128, B2b, nullptr, eb2, 128, NE, ET, 128, 128, 128, 1, 1.f, EB);
  gemm(eb2, 128, B3t, 128, B3b, lg, nullptr, HEADS, NE, ET, HEADS, 16, 128, 0, isd, EB);

  // ---- 3: value path: gelu(gelu(M1[:,128:]@V1)@V2)@V3 -> Vm f32 ----
  gemm(M1 + 128, 384, V1t, 256, V1b, nullptr, eb1, 128, NE, ET, 128, 128, 256, 2, 1.f, EB);
  gemm(eb1, 128, V2t, 128, V2b, nullptr, eb2, 128, NE, ET, 128, 128, 128, 2, 1.f, EB);
  gemm(eb2, 128, V3t, 128, V3b, Vm, nullptr, 128, NE, ET, 128, 128, 128, 0, 1.f, EB);

  // ---- 4: segment softmax over src + weighted scatter ----
  fill_f32_k<<<blks((long)NN*HEADS), 256, 0, stream>>>(mx, -3.0e38f, (long)NN*HEADS);
  fill_f32_k<<<blks((long)NN*HEADS), 256, 0, stream>>>(den, 0.f, (long)NN*HEADS);
  fill_f32_k<<<blks((long)NPAD*H), 256, 0, stream>>>(agg, 0.f, (long)NPAD*H);
  seg_max_k<<<blks((long)NE*HEADS), 256, 0, stream>>>(lg, src, mx, NE*HEADS);
  seg_exp_k<<<blks((long)NE*HEADS), 256, 0, stream>>>(lg, src, mx, den, NE*HEADS);
  seg_agg_k<<<blks((long)NE*H), 256, 0, stream>>>(lg, den, Vm, src, agg, (long)NE*H);

  // ---- 5: dh = agg @ W_O ; BN0 ----
  cvt_bf_k<<<blks((long)NPAD*H), 256, 0, stream>>>(agg, aggb, (long)NPAD*H);
  gemm(aggb, 128, WOt, 128, nullptr, dh, nullptr, 128, NN, NT_, 128, 128, 128, 0, 1.f, NB);
  bn_reduce_k<<<H, 256, 0, stream>>>(h_V, dh, NN, mv0);
  bn_norm_k<<<blks((long)NN*H), 256, 0, stream>>>(h_V, dh, mv0, N0g, N0b, hVa, hVab, (long)NN*H);

  // ---- 6: dense block + BN1 -> hV (out) ----
  gemm(hVab, 128, D1t, 128, D1b, nullptr, d1b_, 512, NN, NT_, 512, 512, 128, 1, 1.f, NB);
  gemm(d1b_, 512, D2t, 512, D2b, dh, nullptr, 128, NN, NT_, 128, 128, 512, 0, 1.f, NB);
  bn_reduce_k<<<H, 256, 0, stream>>>(hVa, dh, NN, mv1);
  bn_norm_k<<<blks((long)NN*H), 256, 0, stream>>>(hVa, dh, mv1, N1g, N1b, outV, nullptr, (long)NN*H);

  // ---- 7: edge update: [hV[src], h_E, hV[dst]] -> MLP -> msg ; BN2 -> hE (out) ----
  build_edge3_k<<<blks((long)NE*384), 256, 0, stream>>>(outV, h_E, src, dst, M1, (long)NE*384);
  gemm(M1, 384, W11t, 384, W11b, nullptr, eb1, 128, NE, ET, 128, 128, 384, 2, 1.f, EB);
  gemm(eb1, 128, W12t, 128, W12b, nullptr, eb2, 128, NE, ET, 128, 128, 128, 2, 1.f, EB);
  gemm(eb2, 128, W13t, 128, W13b, Vm, nullptr, 128, NE, ET, 128, 128, 128, 0, 1.f, EB);
  bn_reduce_k<<<H, 256, 0, stream>>>(h_E, Vm, NE, mv2);
  bn_norm_k<<<blks((long)NE*H), 256, 0, stream>>>(h_E, Vm, mv2, N2g, N2b, outE, nullptr, (long)NE*H);
}
